// CompressedLinear_39822936768839
// MI455X (gfx1250) — matrix operand order repaired and checked by reading, not compiled, not run on hardware
//
#include <hip/hip_runtime.h>

// ---------------------------------------------------------------- constants
constexpr int NTOK = 8192;
constexpr int DIN  = 4096;
constexpr int DOUT = 4096;
constexpr int RANK = 64;

// LDS tile row stride in ushorts: 32 data + 8 pad -> 80B rows.
// 80B = 20 dwords; 20*k mod 64 gives disjoint 4-bank groups for 16 lanes,
// so per-lane b128 fragment reads are bank-conflict-free, and every row
// start stays 16B aligned.
constexpr int LDS_STRIDE = 40;

// workspace layout (bytes)
constexpr size_t XB_OFF = 0;                                   // bf16 x      [NTOK][DIN]
constexpr size_t XB_SZ  = (size_t)NTOK * DIN * 2;              // 64 MB
constexpr size_t WB_OFF = XB_OFF + XB_SZ;                      // bf16 Q      [DOUT][DIN]
constexpr size_t WB_SZ  = (size_t)DOUT * DIN * 2;              // 32 MB
constexpr size_t UB_OFF = WB_OFF + WB_SZ;                      // bf16 U      [DOUT][RANK]
constexpr size_t UB_SZ  = (size_t)DOUT * RANK * 2;
constexpr size_t DB_OFF = UB_OFF + UB_SZ;                      // bf16 D      [RANK][DIN]
constexpr size_t DB_SZ  = (size_t)RANK * DIN * 2;
constexpr size_t TB_OFF = DB_OFF + DB_SZ;                      // bf16 T=xD^T [NTOK][RANK]

// ---------------------------------------------------------------- types
typedef __attribute__((ext_vector_type(16))) __bf16        v16bf;
typedef __attribute__((ext_vector_type(8)))  float         v8f;
typedef __attribute__((ext_vector_type(4)))  unsigned int  u32x4;
typedef __attribute__((ext_vector_type(4)))  float         f32x4;
typedef __attribute__((ext_vector_type(4)))  int           i32x4;

typedef __attribute__((address_space(3))) void       as3_void;
typedef __attribute__((address_space(1))) const void as1_cvoid;

// exact parameter types of the async-copy builtin (from the round-2 diagnostic:
// "int __vector(4) __device__ *"): GCC-style 16B int vector, AS1 / AS3 pointers
typedef int gcc_v4i __attribute__((__vector_size__(4 * sizeof(int))));
typedef __attribute__((address_space(1))) gcc_v4i as1_v4i;
typedef __attribute__((address_space(3))) gcc_v4i as3_v4i;

// ---- CDNA5 async global->LDS path (guarded; sync fallback otherwise) ----
#if defined(__has_builtin)
#if __has_builtin(__builtin_amdgcn_global_load_async_to_lds_b128)
#define HAVE_ASYNC_CP 1
#endif
#endif
#ifndef HAVE_ASYNC_CP
#define HAVE_ASYNC_CP 0
#endif

#if HAVE_ASYNC_CP
#if defined(__has_builtin) && __has_builtin(__builtin_amdgcn_s_wait_asynccnt)
#define WAIT_ASYNC() __builtin_amdgcn_s_wait_asynccnt(0)
#else
#define WAIT_ASYNC() asm volatile("s_wait_asynccnt 0x0" ::: "memory")
#endif
__device__ __forceinline__ void cp16_async(unsigned short* lds_dst,
                                           const unsigned short* g_src) {
  __builtin_amdgcn_global_load_async_to_lds_b128(
      (as1_v4i*)(as1_cvoid*)g_src,
      (as3_v4i*)(as3_void*)lds_dst, 0, 0);
}
#endif

struct FragBits { u32x4 lo, hi; };   // 32 bytes == sizeof(v16bf)

__device__ __forceinline__ unsigned short f2bf(float f) {
  unsigned int u = __float_as_uint(f);
  unsigned int r = u + 0x7FFFu + ((u >> 16) & 1u);   // round-to-nearest-even
  return (unsigned short)(r >> 16);
}
__device__ __forceinline__ unsigned int pack2(float a, float b) {
  return (unsigned int)f2bf(a) | ((unsigned int)f2bf(b) << 16);
}
__device__ __forceinline__ v16bf load_frag(const unsigned short* lo,
                                           const unsigned short* hi) {
  FragBits fb;
  fb.lo = *reinterpret_cast<const u32x4*>(lo);
  fb.hi = *reinterpret_cast<const u32x4*>(hi);
  return __builtin_bit_cast(v16bf, fb);
}

// ---------------------------------------------------------------- converters
__global__ void k_f32_to_bf16(const float* __restrict__ src,
                              unsigned short* __restrict__ dst, int n8) {
  int i = blockIdx.x * blockDim.x + threadIdx.x;
  if (i >= n8) return;
  const f32x4* s = reinterpret_cast<const f32x4*>(src) + (size_t)i * 2;
  f32x4 a = s[0], b = s[1];
  u32x4 o;
  o.x = pack2(a.x, a.y); o.y = pack2(a.z, a.w);
  o.z = pack2(b.x, b.y); o.w = pack2(b.z, b.w);
  reinterpret_cast<u32x4*>(dst)[i] = o;
}

__global__ void k_i32_to_bf16(const int* __restrict__ src,
                              unsigned short* __restrict__ dst, int n8) {
  int i = blockIdx.x * blockDim.x + threadIdx.x;
  if (i >= n8) return;
  const i32x4* s = reinterpret_cast<const i32x4*>(src) + (size_t)i * 2;
  i32x4 a = s[0], b = s[1];
  u32x4 o;
  o.x = pack2((float)a.x, (float)a.y); o.y = pack2((float)a.z, (float)a.w);
  o.z = pack2((float)b.x, (float)b.y); o.w = pack2((float)b.z, (float)b.w);
  reinterpret_cast<u32x4*>(dst)[i] = o;
}

// ---------------------------------------------------------------- T = x @ D^T
__global__ __launch_bounds__(256)
void k_lowrank_t(const unsigned short* __restrict__ xb,
                 const unsigned short* __restrict__ db,
                 unsigned short* __restrict__ tb) {
  const int wave = threadIdx.x >> 5;
  const int lane = threadIdx.x & 31;
  const int tile = blockIdx.x * 8 + wave;
  const int row0 = (tile >> 2) << 4;
  const int col0 = (tile & 3)  << 4;
  const int lm = lane & 15;
  const int hh = lane >> 4;

  const unsigned short* ap = xb + (size_t)(row0 + lm) * DIN + hh * 8;
  const unsigned short* bp = db + (size_t)(col0 + lm) * DIN + hh * 8;
  v8f acc = {};
  for (int k0 = 0; k0 < DIN; k0 += 32) {
    v16bf a = load_frag(ap + k0, ap + k0 + 16);
    v16bf b = load_frag(bp + k0, bp + k0 + 16);
    acc = __builtin_amdgcn_wmma_f32_16x16x32_bf16(false, a, false, b,
                                                  (short)0, acc, false, false);
  }
  const int orow = row0 + hh * 8;
  const int ocol = col0 + lm;
#pragma unroll
  for (int r = 0; r < 8; ++r)
    tb[(size_t)(orow + r) * RANK + ocol] = f2bf(acc[r]);
}

// ---------------------------------------------------------------- main GEMM
__device__ __forceinline__ void mma_stage(const unsigned short* __restrict__ A,
                                          const unsigned short* __restrict__ B,
                                          int waveM, int waveN, int lm, int hh,
                                          v8f (&acc)[2][4]) {
  v16bf afr[2], bfr[4];
#pragma unroll
  for (int mi = 0; mi < 2; ++mi) {
    const unsigned short* p = &A[(waveM * 32 + mi * 16 + lm) * LDS_STRIDE + hh * 8];
    afr[mi] = load_frag(p, p + 16);
  }
#pragma unroll
  for (int ni = 0; ni < 4; ++ni) {
    const unsigned short* p = &B[(waveN * 64 + ni * 16 + lm) * LDS_STRIDE + hh * 8];
    bfr[ni] = load_frag(p, p + 16);
  }
#pragma unroll
  for (int mi = 0; mi < 2; ++mi)
#pragma unroll
    for (int ni = 0; ni < 4; ++ni)
      acc[mi][ni] = __builtin_amdgcn_wmma_f32_16x16x32_bf16(
          false, afr[mi], false, bfr[ni], (short)0, acc[mi][ni], false, false);
}

__global__ __launch_bounds__(256)
void k_main_gemm(const unsigned short* __restrict__ xb,
                 const unsigned short* __restrict__ wb,
                 const unsigned short* __restrict__ ub,
                 const unsigned short* __restrict__ tb,
                 const float* __restrict__ scales,
                 float* __restrict__ out) {
  __shared__ __align__(16) unsigned short ldsA[2][128 * LDS_STRIDE];
  __shared__ __align__(16) unsigned short ldsB[2][128 * LDS_STRIDE];

  const int t     = threadIdx.x;
  const int wave  = t >> 5;
  const int lane  = t & 31;
  const int waveM = wave & 3;
  const int waveN = wave >> 2;
  const int lm    = lane & 15;
  const int hh    = lane >> 4;

  const int rowBase = blockIdx.y * 128;
  const int colBase = blockIdx.x * 128;

  const int ldRow = t >> 2;
  const int ldOff = (t & 3) * 8;
  const unsigned short* aSrc0 = xb + (size_t)(rowBase +      ldRow) * DIN + ldOff;
  const unsigned short* aSrc1 = xb + (size_t)(rowBase + 64 + ldRow) * DIN + ldOff;
  const unsigned short* bSrc0 = wb + (size_t)(colBase +      ldRow) * DIN + ldOff;
  const unsigned short* bSrc1 = wb + (size_t)(colBase + 64 + ldRow) * DIN + ldOff;
  const int li0 = (ldRow)      * LDS_STRIDE + ldOff;
  const int li1 = (ldRow + 64) * LDS_STRIDE + ldOff;

  v8f acc[2][4] = {};
  constexpr int NK = DIN / 32;

#if HAVE_ASYNC_CP
  // ---- CDNA5 path: async DMA global->LDS, ASYNCcnt-tracked, double-buffered
  cp16_async(&ldsA[0][li0], aSrc0);
  cp16_async(&ldsA[0][li1], aSrc1);
  cp16_async(&ldsB[0][li0], bSrc0);
  cp16_async(&ldsB[0][li1], bSrc1);
  WAIT_ASYNC();
  __syncthreads();

#pragma unroll 1
  for (int kt = 0; kt < NK; ++kt) {
    const int cur = kt & 1;
    if (kt + 1 < NK) {               // prefetch next slab into other buffer
      const int k0 = (kt + 1) * 32;
      cp16_async(&ldsA[cur ^ 1][li0], aSrc0 + k0);
      cp16_async(&ldsA[cur ^ 1][li1], aSrc1 + k0);
      cp16_async(&ldsB[cur ^ 1][li0], bSrc0 + k0);
      cp16_async(&ldsB[cur ^ 1][li1], bSrc1 + k0);
    }
    mma_stage(ldsA[cur], ldsB[cur], waveM, waveN, lm, hh, acc);  // overlaps DMA
    WAIT_ASYNC();                    // own async copies landed in LDS
    __syncthreads();                 // everyone's copies visible; reads done
  }
#else
  // ---- fallback: synchronous staging through VGPRs
#pragma unroll 1
  for (int kt = 0; kt < NK; ++kt) {
    const int k0 = kt * 32;
    u32x4 ra0 = *reinterpret_cast<const u32x4*>(aSrc0 + k0);
    u32x4 ra1 = *reinterpret_cast<const u32x4*>(aSrc1 + k0);
    u32x4 rb0 = *reinterpret_cast<const u32x4*>(bSrc0 + k0);
    u32x4 rb1 = *reinterpret_cast<const u32x4*>(bSrc1 + k0);
    __syncthreads();
    *reinterpret_cast<u32x4*>(&ldsA[0][li0]) = ra0;
    *reinterpret_cast<u32x4*>(&ldsA[0][li1]) = ra1;
    *reinterpret_cast<u32x4*>(&ldsB[0][li0]) = rb0;
    *reinterpret_cast<u32x4*>(&ldsB[0][li1]) = rb1;
    __syncthreads();
    mma_stage(ldsA[0], ldsB[0], waveM, waveN, lm, hh, acc);
  }
#endif

  // epilogue 1: per-output-channel dequant scale
#pragma unroll
  for (int ni = 0; ni < 4; ++ni) {
    float s = scales[colBase + waveN * 64 + ni * 16 + lm];
#pragma unroll
    for (int mi = 0; mi < 2; ++mi)
      acc[mi][ni] = acc[mi][ni] * s;
  }

  // epilogue 2: low-rank adapter  T @ U^T  (K = 64, straight from L2)
#pragma unroll
  for (int k0 = 0; k0 < RANK; k0 += 32) {
    v16bf at[2], bt[4];
#pragma unroll
    for (int mi = 0; mi < 2; ++mi) {
      const unsigned short* p =
          tb + (size_t)(rowBase + waveM * 32 + mi * 16 + lm) * RANK + k0 + hh * 8;
      at[mi] = load_frag(p, p + 16);
    }
#pragma unroll
    for (int ni = 0; ni < 4; ++ni) {
      const unsigned short* p =
          ub + (size_t)(colBase + waveN * 64 + ni * 16 + lm) * RANK + k0 + hh * 8;
      bt[ni] = load_frag(p, p + 16);
    }
#pragma unroll
    for (int mi = 0; mi < 2; ++mi)
#pragma unroll
      for (int ni = 0; ni < 4; ++ni)
        acc[mi][ni] = __builtin_amdgcn_wmma_f32_16x16x32_bf16(
            false, at[mi], false, bt[ni], (short)0, acc[mi][ni], false, false);
  }

  // store (coalesced 64B per lane half-group)
#pragma unroll
  for (int mi = 0; mi < 2; ++mi)
#pragma unroll
    for (int ni = 0; ni < 4; ++ni) {
      const int row = rowBase + waveM * 32 + mi * 16 + hh * 8;
      const int col = colBase + waveN * 64 + ni * 16 + lm;
      float* op = out + (size_t)row * DOUT + col;
#pragma unroll
      for (int r = 0; r < 8; ++r)
        op[(size_t)r * DOUT] = acc[mi][ni][r];
    }
}

// ---------------------------------------------------------------- launch
extern "C" void kernel_launch(void* const* d_in, const int* in_sizes, int n_in,
                              void* d_out, int out_size, void* d_ws, size_t ws_size,
                              hipStream_t stream) {
  const float* x      = (const float*)d_in[0];
  const float* scales = (const float*)d_in[1];
  const float* U      = (const float*)d_in[2];
  const float* D      = (const float*)d_in[3];
  const int*   Q      = (const int*)  d_in[4];
  float* out = (float*)d_out;

  char* ws = (char*)d_ws;
  unsigned short* xb = (unsigned short*)(ws + XB_OFF);
  unsigned short* wb = (unsigned short*)(ws + WB_OFF);
  unsigned short* ub = (unsigned short*)(ws + UB_OFF);
  unsigned short* db = (unsigned short*)(ws + DB_OFF);
  unsigned short* tb = (unsigned short*)(ws + TB_OFF);

  {
    int n8 = NTOK * DIN / 8;
    k_f32_to_bf16<<<(n8 + 255) / 256, 256, 0, stream>>>(x, xb, n8);
  }
  {
    int n8 = DOUT * DIN / 8;
    k_i32_to_bf16<<<(n8 + 255) / 256, 256, 0, stream>>>(Q, wb, n8);
  }
  {
    int n8 = DOUT * RANK / 8;
    k_f32_to_bf16<<<(n8 + 255) / 256, 256, 0, stream>>>(U, ub, n8);
  }
  {
    int n8 = RANK * DIN / 8;
    k_f32_to_bf16<<<(n8 + 255) / 256, 256, 0, stream>>>(D, db, n8);
  }

  k_lowrank_t<<<256, 256, 0, stream>>>(xb, db, tb);

  k_main_gemm<<<dim3(DOUT / 128, NTOK / 128), 256, 0, stream>>>(
      xb, wb, ub, tb, scales, out);
}
